// TRSModel_46815143526925
// MI455X (gfx1250) — matrix operand order repaired and checked by reading, not compiled, not run on hardware
//
#include <hip/hip_runtime.h>

#define BT 32
#define HH 256
#define TT 128
#define PP 6

typedef __attribute__((ext_vector_type(16))) __bf16 v16bf;
typedef __attribute__((ext_vector_type(8)))  float  v8f;

union FragU { v16bf v; float4 q[2]; };

__device__ __forceinline__ v8f wmma_bf16(v16bf a, v16bf b, v8f c) {
  // (neg_a, A, neg_b, B, c_mod, C, reuse_a, reuse_b)
  return __builtin_amdgcn_wmma_f32_16x16x32_bf16(false, a, false, b, (short)0, c, false, false);
}

// A fragment (16x32, bf16) from a row-major [rows x 256] tile.
// Lane (m = lane&15): VGPR0-3 hold K=kbase..kbase+7, VGPR4-7 hold K=kbase+16..kbase+23,
// kbase = 8*(lane>>4)  -> two contiguous 16B chunks per lane.
__device__ __forceinline__ v16bf load_a_frag(const __bf16* base, int mb, int kb, int lane) {
  const int m = mb * 16 + (lane & 15);
  const int k = kb * 32 + ((lane >> 4) << 3);
  const __bf16* p = base + m * HH + k;
  FragU f;
  f.q[0] = *reinterpret_cast<const float4*>(p);
  f.q[1] = *reinterpret_cast<const float4*>(p + 16);
  return f.v;
}

// B fragment (32x16 = U.T chunk) from row-major U[256x256] (bf16).
// Lane (n = lane&15, kbase = 8*(lane>>4)): holds U[n, kb*32 + kbase .. +8) and U[n, kb*32 + kbase + 16 .. +8).
__device__ __forceinline__ v16bf load_b_frag(const __bf16* U, int nb, int kb, int lane) {
  const int n = nb * 16 + (lane & 15);
  const int k = kb * 32 + ((lane >> 4) << 3);
  const __bf16* p = U + n * HH + k;
  FragU f;
  f.q[0] = *reinterpret_cast<const float4*>(p);
  f.q[1] = *reinterpret_cast<const float4*>(p + 16);
  return f.v;
}

__device__ __forceinline__ float sigm(float x)   { return 1.f / (1.f + __expf(-x)); }
__device__ __forceinline__ float tanh_f(float x) { float t = __expf(2.f * x); return 1.f - 2.f / (t + 1.f); }
__device__ __forceinline__ float sel3(float a, float b, float c, int s) {
  return s == 0 ? a : (s == 1 ? b : c);
}
__device__ __forceinline__ v8f vzero() {
  v8f z;
#pragma unroll
  for (int i = 0; i < 8; ++i) z[i] = 0.f;
  return z;
}

// acc[mb] (mb=0,1) accumulate A(32x256 from As) * B.T for this wave's single n-block.
// kb unroll limited to 2 so at most 4 B-fragments are in flight -> no spills.
__device__ __forceinline__ void gemm2(const __bf16* As, const __bf16* B0, const __bf16* B1,
                                      v8f acc0[2], v8f acc1[2], int nb, int lane) {
#pragma unroll 2
  for (int kb = 0; kb < 8; ++kb) {
    v16bf a0 = load_a_frag(As, 0, kb, lane);
    v16bf a1 = load_a_frag(As, 1, kb, lane);
    v16bf b0 = load_b_frag(B0, nb, kb, lane);
    v16bf b1 = load_b_frag(B1, nb, kb, lane);
    acc0[0] = wmma_bf16(a0, b0, acc0[0]);
    acc0[1] = wmma_bf16(a1, b0, acc0[1]);
    acc1[0] = wmma_bf16(a0, b1, acc1[0]);
    acc1[1] = wmma_bf16(a1, b1, acc1[1]);
  }
}

__device__ __forceinline__ void gemm1(const __bf16* As, const __bf16* B0,
                                      v8f acc0[2], int nb, int lane) {
#pragma unroll 2
  for (int kb = 0; kb < 8; ++kb) {
    v16bf a0 = load_a_frag(As, 0, kb, lane);
    v16bf a1 = load_a_frag(As, 1, kb, lane);
    v16bf b0 = load_b_frag(B0, nb, kb, lane);
    acc0[0] = wmma_bf16(a0, b0, acc0[0]);
    acc0[1] = wmma_bf16(a1, b0, acc0[1]);
  }
}

// ---------------------------------------------------------------------------
// Main fused 2-layer GRU kernel: one workgroup = 32 batch rows, all 128 steps.
// 512 threads = 16 waves; wave w owns output columns [16w, 16w+16) x all 32 rows.
// ---------------------------------------------------------------------------
extern "C" __global__ __launch_bounds__(512, 1) void rnn_main(
    const float* __restrict__ x,
    const __bf16* __restrict__ Wp,      // 9 x [256x256] bf16: Uz0,Ur0,Uh0, Wz1h,Wr1h,Wh1h, Uz1,Ur1,Uh1
    const float* __restrict__ tabsG,    // 8 tables x [3][256]
    const float* __restrict__ col0G,    // 3 x [256]
    const float* __restrict__ lamsG,    // 6 x [256]
    const float* __restrict__ fc_w,
    const float* __restrict__ fc_b,
    float* __restrict__ out)
{
  extern __shared__ char smem[];
  __bf16* h0s = reinterpret_cast<__bf16*>(smem + 0);        // 16384
  __bf16* h1s = reinterpret_cast<__bf16*>(smem + 16384);    // 16384
  __bf16* rhs = reinterpret_cast<__bf16*>(smem + 32768);    // 16384 (r*h scratch)
  float*  mps = reinterpret_cast<float*>(smem + 49152);     // 16384 map[row][t]
  unsigned char* ids = reinterpret_cast<unsigned char*>(smem + 65536); // 4096 idx[row][t]
  float*  tbs = reinterpret_cast<float*>(smem + 69632);     // 24576: 8x[3][256]
  float*  c0s = reinterpret_cast<float*>(smem + 94208);     // 3072
  float*  lms = reinterpret_cast<float*>(smem + 97280);     // 6144
  float*  h1f = reinterpret_cast<float*>(smem + 32768);     // aliases rhs+mps, used post-loop

  const int tid  = threadIdx.x;
  const int lane = tid & 31;
  const int wid  = tid >> 5;     // 0..15 -> n-block
  const int b0   = blockIdx.x * BT;

  // init LDS: zero h states, stage x tile, stage tables
  for (int i = tid; i < BT * HH / 2; i += 512) {
    reinterpret_cast<unsigned int*>(h0s)[i] = 0u;
    reinterpret_cast<unsigned int*>(h1s)[i] = 0u;
  }
  for (int i = tid; i < BT * TT; i += 512) {
    const int r = i >> 7, t = i & 127;
    const float* px = x + (((b0 + r) * TT) + t) * 2;
    mps[i] = px[0];
    int c = (int)px[1];
    c = c < 0 ? 0 : (c > 2 ? 2 : c);
    ids[i] = (unsigned char)c;
  }
  for (int i = tid; i < 8 * 3 * HH; i += 512) tbs[i] = tabsG[i];
  for (int i = tid; i < 3 * HH; i += 512) c0s[i] = col0G[i];
  for (int i = tid; i < 6 * HH; i += 512) lms[i] = lamsG[i];
  __syncthreads();

  const __bf16* Uz0  = Wp + 0 * 65536;
  const __bf16* Ur0  = Wp + 1 * 65536;
  const __bf16* Uh0  = Wp + 2 * 65536;
  const __bf16* Wz1h = Wp + 3 * 65536;
  const __bf16* Wr1h = Wp + 4 * 65536;
  const __bf16* Wh1h = Wp + 5 * 65536;
  const __bf16* Uz1  = Wp + 6 * 65536;
  const __bf16* Ur1  = Wp + 7 * 65536;
  const __bf16* Uh1  = Wp + 8 * 65536;

  const int hi8 = (lane >> 4) << 3;   // C/D layout: high lanes -> rows +8
  const int ncl = lane & 15;
  const int n   = wid * 16 + ncl;     // this wave's output column (fixed)

  // Hoist n-dependent per-lane constants out of the time loop (9 scalars).
  const float c0z = c0s[n],          c0r = c0s[HH + n],      c0h = c0s[2 * HH + n];
  const float lz  = lms[n],          lr  = lms[HH + n],      lh  = lms[2 * HH + n];
  const float lz1 = lms[3 * HH + n], lr1 = lms[4 * HH + n],  lh1 = lms[5 * HH + n];

  // fp32 h-state held in registers for the wave's 32x16 region
  v8f h0r[2], h1r[2];
  h0r[0] = vzero(); h0r[1] = vzero(); h1r[0] = vzero(); h1r[1] = vzero();

#pragma unroll 1
  for (int t = 0; t < TT; ++t) {
    // ================= layer 0: z, r gates =================
    v8f az[2], ar[2];
    az[0] = vzero(); az[1] = vzero(); ar[0] = vzero(); ar[1] = vzero();
    gemm2(h0s, Uz0, Ur0, az, ar, wid, lane);

    {
      const float tzA = tbs[0 * 768 + n], tzB = tbs[0 * 768 + 256 + n], tzC = tbs[0 * 768 + 512 + n];
      const float trA = tbs[1 * 768 + n], trB = tbs[1 * 768 + 256 + n], trC = tbs[1 * 768 + 512 + n];
      const float tiA = tbs[3 * 768 + n], tiB = tbs[3 * 768 + 256 + n], tiC = tbs[3 * 768 + 512 + n];
#pragma unroll
      for (int mb = 0; mb < 2; ++mb)
#pragma unroll
        for (int i = 0; i < 8; ++i) {
          const int row = mb * 16 + hi8 + i;
          const int c   = ids[row * TT + t];
          const float mv = mps[row * TT + t];
          const float ti = sel3(tiA, tiB, tiC, c);
          const float zv = sigm(az[mb][i] + c0z * mv + sel3(tzA, tzB, tzC, c) + lz * ti);
          const float rv = sigm(ar[mb][i] + c0r * mv + sel3(trA, trB, trC, c) + lr * ti);
          az[mb][i] = zv;                                   // keep z for update
          rhs[row * HH + n] = (__bf16)(rv * h0r[mb][i]);    // (r*h0) for Uh0 GEMM
        }
    }
    __syncthreads();

    // ================= layer 0: candidate + h0 update =================
    v8f ah[2];
    ah[0] = vzero(); ah[1] = vzero();
    gemm1(rhs, Uh0, ah, wid, lane);

    {
      const float thA = tbs[2 * 768 + n], thB = tbs[2 * 768 + 256 + n], thC = tbs[2 * 768 + 512 + n];
      const float tiA = tbs[3 * 768 + n], tiB = tbs[3 * 768 + 256 + n], tiC = tbs[3 * 768 + 512 + n];
#pragma unroll
      for (int mb = 0; mb < 2; ++mb)
#pragma unroll
        for (int i = 0; i < 8; ++i) {
          const int row = mb * 16 + hi8 + i;
          const int c   = ids[row * TT + t];
          const float mv = mps[row * TT + t];
          const float hv = tanh_f(ah[mb][i] + c0h * mv + sel3(thA, thB, thC, c)
                                  + lh * sel3(tiA, tiB, tiC, c));
          const float zv = az[mb][i];
          const float hn = (1.f - zv) * h0r[mb][i] + zv * hv;
          h0r[mb][i] = hn;
          h0s[row * HH + n] = (__bf16)hn;
        }
    }
    __syncthreads();

    // ================= layer 1: z1, r1 gates =================
    az[0] = vzero(); az[1] = vzero(); ar[0] = vzero(); ar[1] = vzero();
    gemm2(h0s, Wz1h, Wr1h, az, ar, wid, lane);  // s1 h-part
    gemm2(h1s, Uz1,  Ur1,  az, ar, wid, lane);  // recurrent part

    {
      const float tzA = tbs[4 * 768 + n], tzB = tbs[4 * 768 + 256 + n], tzC = tbs[4 * 768 + 512 + n];
      const float trA = tbs[5 * 768 + n], trB = tbs[5 * 768 + 256 + n], trC = tbs[5 * 768 + 512 + n];
      const float tiA = tbs[7 * 768 + n], tiB = tbs[7 * 768 + 256 + n], tiC = tbs[7 * 768 + 512 + n];
#pragma unroll
      for (int mb = 0; mb < 2; ++mb)
#pragma unroll
        for (int i = 0; i < 8; ++i) {
          const int row = mb * 16 + hi8 + i;
          const int c   = ids[row * TT + t];
          const float ti = sel3(tiA, tiB, tiC, c);
          const float z1 = sigm(az[mb][i] + sel3(tzA, tzB, tzC, c) + lz1 * ti);
          const float r1 = sigm(ar[mb][i] + sel3(trA, trB, trC, c) + lr1 * ti);
          az[mb][i] = z1;
          rhs[row * HH + n] = (__bf16)(r1 * h1r[mb][i]);    // (r1*h1) for Uh1 GEMM
        }
    }
    __syncthreads();

    // ================= layer 1: candidate + h1 update =================
    ah[0] = vzero(); ah[1] = vzero();
    gemm1(h0s, Wh1h, ah, wid, lane);
    gemm1(rhs, Uh1,  ah, wid, lane);

    {
      const float thA = tbs[6 * 768 + n], thB = tbs[6 * 768 + 256 + n], thC = tbs[6 * 768 + 512 + n];
      const float tiA = tbs[7 * 768 + n], tiB = tbs[7 * 768 + 256 + n], tiC = tbs[7 * 768 + 512 + n];
#pragma unroll
      for (int mb = 0; mb < 2; ++mb)
#pragma unroll
        for (int i = 0; i < 8; ++i) {
          const int row = mb * 16 + hi8 + i;
          const int c   = ids[row * TT + t];
          const float hv = tanh_f(ah[mb][i] + sel3(thA, thB, thC, c)
                                  + lh1 * sel3(tiA, tiB, tiC, c));
          const float z1 = az[mb][i];
          const float hn = (1.f - z1) * h1r[mb][i] + z1 * hv;
          h1r[mb][i] = hn;
          h1s[row * HH + n] = (__bf16)hn;
        }
    }
    __syncthreads();
  }

  // ================= final FC: out = h1 @ fc_w.T + fc_b =================
#pragma unroll
  for (int mb = 0; mb < 2; ++mb)
#pragma unroll
    for (int i = 0; i < 8; ++i)
      h1f[(mb * 16 + hi8 + i) * HH + n] = h1r[mb][i];
  __syncthreads();
  if (tid < BT * PP) {
    const int row = tid / PP, p = tid % PP;
    float s = fc_b[p];
    for (int h = 0; h < HH; ++h) s += h1f[row * HH + h] * fc_w[p * HH + h];
    out[(b0 + row) * PP + p] = s;
  }
}

// ---------------------------------------------------------------------------
// Prep kernel 1: repack the 9 recurrent 256x256 (or 256-col slice) weight
// matrices to bf16, row-major stride 256. These then live in L2 for all steps.
// ---------------------------------------------------------------------------
extern "C" __global__ void pack_weights(
    const float* __restrict__ Uz0, const float* __restrict__ Ur0, const float* __restrict__ Uh0,
    const float* __restrict__ Wz1, const float* __restrict__ Wr1, const float* __restrict__ Wh1,
    const float* __restrict__ Uz1, const float* __restrict__ Ur1, const float* __restrict__ Uh1,
    __bf16* __restrict__ outp)
{
  const int i = blockIdx.x * blockDim.x + threadIdx.x;
  if (i >= 9 * 65536) return;
  const int m = i >> 16;
  const int r = (i >> 8) & 255;
  const int k = i & 255;
  const float* src; int ld;
  switch (m) {
    case 0: src = Uz0; ld = 256; break;
    case 1: src = Ur0; ld = 256; break;
    case 2: src = Uh0; ld = 256; break;
    case 3: src = Wz1; ld = 264; break;   // h-part = cols 0..255 of (256 x 264)
    case 4: src = Wr1; ld = 264; break;
    case 5: src = Wh1; ld = 264; break;
    case 6: src = Uz1; ld = 256; break;
    case 7: src = Ur1; ld = 256; break;
    default: src = Uh1; ld = 256; break;
  }
  outp[i] = (__bf16)src[r * ld + k];
}

// ---------------------------------------------------------------------------
// Prep kernel 2: per-treatment gate tables (K=8/9 GEMMs collapsed since the
// embedding has only NT=3 distinct rows), plus column-0 vectors and lambdas.
// tabs order: Tz0,Tr0,Th0, Tiv0(+Vb0), Tz1a,Tr1a,Th1a, Tiv1(+Vb1)  each [3][256]
// ---------------------------------------------------------------------------
extern "C" __global__ void prep_tables(
    const float* __restrict__ emb,
    const float* __restrict__ Wz0, const float* __restrict__ Wr0, const float* __restrict__ Wh0,
    const float* __restrict__ Vw0, const float* __restrict__ Vb0,
    const float* __restrict__ lz0, const float* __restrict__ lr0, const float* __restrict__ lh0,
    const float* __restrict__ Wz1, const float* __restrict__ Wr1, const float* __restrict__ Wh1,
    const float* __restrict__ Vw1, const float* __restrict__ Vb1,
    const float* __restrict__ lz1, const float* __restrict__ lr1, const float* __restrict__ lh1,
    float* __restrict__ tabs, float* __restrict__ col0, float* __restrict__ lams)
{
  const int i = blockIdx.x * blockDim.x + threadIdx.x;
  if (i < 8 * 3 * 256) {
    const int tb = i / 768;
    const int c  = (i / 256) % 3;
    const int h  = i % 256;
    float s = 0.f;
    switch (tb) {
      case 0: for (int e = 0; e < 8; ++e) s += Wz0[h * 9 + 1 + e] * emb[c * 8 + e]; break;
      case 1: for (int e = 0; e < 8; ++e) s += Wr0[h * 9 + 1 + e] * emb[c * 8 + e]; break;
      case 2: for (int e = 0; e < 8; ++e) s += Wh0[h * 9 + 1 + e] * emb[c * 8 + e]; break;
      case 3: for (int e = 0; e < 8; ++e) s += Vw0[h * 8 + e] * emb[c * 8 + e]; s += Vb0[h]; break;
      case 4: for (int e = 0; e < 8; ++e) s += Wz1[h * 264 + 256 + e] * emb[c * 8 + e]; break;
      case 5: for (int e = 0; e < 8; ++e) s += Wr1[h * 264 + 256 + e] * emb[c * 8 + e]; break;
      case 6: for (int e = 0; e < 8; ++e) s += Wh1[h * 264 + 256 + e] * emb[c * 8 + e]; break;
      default: for (int e = 0; e < 8; ++e) s += Vw1[h * 8 + e] * emb[c * 8 + e]; s += Vb1[h]; break;
    }
    tabs[i] = s;
  } else if (i < 8 * 768 + 3 * 256) {
    const int j = i - 8 * 768;
    const int w = j / 256, h = j % 256;
    const float* src = (w == 0) ? Wz0 : (w == 1) ? Wr0 : Wh0;
    col0[j] = src[h * 9 + 0];
  } else if (i < 8 * 768 + 3 * 256 + 6 * 256) {
    const int j = i - 8 * 768 - 3 * 256;
    const int w = j / 256, h = j % 256;
    const float* raw = (w == 0) ? lz0 : (w == 1) ? lr0 : (w == 2) ? lh0
                     : (w == 3) ? lz1 : (w == 4) ? lr1 : lh1;
    lams[j] = 0.05f + 0.45f / (1.f + __expf(-raw[h]));
  }
}

// ---------------------------------------------------------------------------
extern "C" void kernel_launch(void* const* d_in, const int* in_sizes, int n_in,
                              void* d_out, int out_size, void* d_ws, size_t ws_size,
                              hipStream_t stream)
{
  const float* x    = (const float*)d_in[0];
  const float* emb  = (const float*)d_in[1];
  const float* Wz0  = (const float*)d_in[2];
  const float* Wr0  = (const float*)d_in[3];
  const float* Wh0  = (const float*)d_in[4];
  const float* Uz0  = (const float*)d_in[5];
  const float* Ur0  = (const float*)d_in[6];
  const float* Uh0  = (const float*)d_in[7];
  const float* Vw0  = (const float*)d_in[8];
  const float* Vb0  = (const float*)d_in[9];
  const float* lz0  = (const float*)d_in[10];
  const float* lr0  = (const float*)d_in[11];
  const float* lh0  = (const float*)d_in[12];
  const float* Wz1  = (const float*)d_in[13];
  const float* Wr1  = (const float*)d_in[14];
  const float* Wh1  = (const float*)d_in[15];
  const float* Uz1  = (const float*)d_in[16];
  const float* Ur1  = (const float*)d_in[17];
  const float* Uh1  = (const float*)d_in[18];
  const float* Vw1  = (const float*)d_in[19];
  const float* Vb1  = (const float*)d_in[20];
  const float* lz1  = (const float*)d_in[21];
  const float* lr1  = (const float*)d_in[22];
  const float* lh1  = (const float*)d_in[23];
  const float* fc_w = (const float*)d_in[24];
  const float* fc_b = (const float*)d_in[25];

  char* ws = (char*)d_ws;
  __bf16* Wp  = (__bf16*)ws;                       // 9*65536*2 = 1,179,648 B
  float* tabs = (float*)(ws + 9 * 65536 * 2);      // 24,576 B
  float* col0 = tabs + 8 * 3 * 256;                // 3,072 B
  float* lams = col0 + 3 * 256;                    // 6,144 B

  pack_weights<<<(9 * 65536) / 256, 256, 0, stream>>>(
      Uz0, Ur0, Uh0, Wz1, Wr1, Wh1, Uz1, Ur1, Uh1, Wp);

  prep_tables<<<(8 * 768 + 3 * 256 + 6 * 256 + 255) / 256, 256, 0, stream>>>(
      emb, Wz0, Wr0, Wh0, Vw0, Vb0, lz0, lr0, lh0,
      Wz1, Wr1, Wh1, Vw1, Vb1, lz1, lr1, lh1,
      tabs, col0, lams);

  const size_t shmem = 103424;  // h0s+h1s+rhs+mps+ids+tabs+col0+lams
  rnn_main<<<2048 / BT, 512, shmem, stream>>>(
      x, Wp, tabs, col0, lams, fc_w, fc_b, (float*)d_out);
}
